// GoalDecoderLSTM_5549097746438
// MI455X (gfx1250) — matrix operand order repaired and checked by reading, not compiled, not run on hardware
//
#include <hip/hip_runtime.h>

typedef __attribute__((ext_vector_type(16))) _Float16 v16h;
typedef __attribute__((ext_vector_type(8)))  _Float16 v8h;
typedef __attribute__((ext_vector_type(8)))  float    v8f;

#define SEQ_LEN_ 30
#define WAVES_   4
#define KTOT_    96   // K space: [0,16)=x, [16,80)=h, 80=bias-one, (80,96)=zero pad

// Storage permutation: within each 32-K chunk the four 8-half blocks are stored
// in order {0,2,1,3}, so lane (col,hlf) reads its 16 A-operand halves as ONE
// contiguous 32B run at byte offset col*192 + ck*64 + hlf*32.
static __device__ __forceinline__ int posOf(int K) {
    const int pm[4] = {0, 2, 1, 3};
    return ((K >> 5) << 5) + pm[(K >> 3) & 3] * 8 + (K & 7);
}

static __device__ __forceinline__ float fast_sig(float x) {
    return __builtin_amdgcn_rcpf(1.0f + __expf(-x));
}
static __device__ __forceinline__ float fast_tanh(float x) {
    float e = __expf(2.0f * x);                       // stable at both extremes
    return 1.0f - 2.0f * __builtin_amdgcn_rcpf(e + 1.0f);
}
static __device__ __forceinline__ float lrelu(float x) {
    return x > 0.0f ? x : 0.01f * x;
}

__global__ __launch_bounds__(128) void goal_lstm_kernel(
    const float* __restrict__ traj_abs, const float* __restrict__ traj_rel,
    const float* __restrict__ h0,       const float* __restrict__ c0,
    const float* __restrict__ goals,
    const float* __restrict__ W_ih,     const float* __restrict__ W_hh,
    const float* __restrict__ b_ih,     const float* __restrict__ b_hh,
    const float* __restrict__ W_se,     const float* __restrict__ b_se,
    const float* __restrict__ W_h2p,    const float* __restrict__ b_h2p,
    const float* __restrict__ W_goal,   const float* __restrict__ b_goal,
    const float* __restrict__ W_abs,    const float* __restrict__ b_abs,
    float* __restrict__ out, int Btot)
{
    // [W_ih | W_hh | bias@K=80] packed as WMMA B-operand fragments (16 N-tiles x 3 K-chunks)
    __shared__ alignas(32) _Float16 wfrag[48 * 32 * 16];          // 48 KB
    __shared__ alignas(32) _Float16 hbuf[WAVES_][16 * KTOT_];     // 12 KB, permuted layout
    __shared__ float W1P[2 * 80];     // W_h2p[:,0:64] permuted to storage order (pos 8..87)
    __shared__ float Mgl[64 * 2];     // W_goal^T @ W_h2p[:,128:192]^T  (goals -> rel)
    __shared__ float Wsel[16 * 2];    // W_se
    __shared__ float bsel[16];        // b_se
    __shared__ float M2l[4];          // W_abs^T @ W_h2p[:,64:128]^T (2x2, abs_pos -> rel)
    __shared__ float crell[2];        // b_h2p + b_abs-term + b_goal-term

    const int tid = threadIdx.x;

    // ---------------- cooperative setup ----------------
    // W1 permuted to the hbuf storage order, covering positions 8..87 (idx 0..79):
    // h positions get the matching W_h2p weight, x/bias/pad positions get 0.
    for (int e = tid; e < 160; e += 128) {
        int d = e / 80, idx = e % 80;
        int pos = idx + 8;
        const int pm[4] = {0, 2, 1, 3};
        int K = ((pos >> 5) << 5) + pm[(pos >> 3) & 3] * 8 + (pos & 7); // perm is an involution
        W1P[d * 80 + idx] = (K >= 16 && K < 80) ? W_h2p[d * 192 + (K - 16)] : 0.0f;
    }
    {                                                 // Mg (all 128 threads, 128 outputs)
        int j = tid >> 1, d = tid & 1;
        float s = 0.f;
        for (int n = 0; n < 64; ++n) s += W_goal[n * 64 + j] * W_h2p[d * 192 + 128 + n];
        Mgl[j * 2 + d] = s;
    }
    if (tid < 32) Wsel[tid] = W_se[tid];
    if (tid < 16) bsel[tid] = b_se[tid];
    if (tid < 4) {
        int c = tid >> 1, d = tid & 1;
        float s = 0.f;
        for (int n = 0; n < 64; ++n) s += W_abs[n * 2 + c] * W_h2p[d * 192 + 64 + n];
        M2l[c * 2 + d] = s;
    }
    if (tid < 2) {
        int d = tid;
        float s = b_h2p[d];
        for (int n = 0; n < 64; ++n) s += b_abs[n]  * W_h2p[d * 192 + 64 + n];
        for (int n = 0; n < 64; ++n) s += b_goal[n] * W_h2p[d * 192 + 128 + n];
        crell[d] = s;
    }
    // Pack weights into v_wmma_f32_16x16x32_f16 B-operand layout:
    // fragment (nt, ck): lane ln holds W[n = 16*nt + (ln&15)][K = 32*ck + 8*(ln>>4) + kk + (kk&8)]
    // K==80 carries the fused bias (pairs with the constant-1 input at A's K=80).
    for (int it = tid; it < 48 * 32; it += 128) {
        int f  = it >> 5, ln2 = it & 31;
        int nt = f / 3,   ck  = f % 3;
        int n  = nt * 16 + (ln2 & 15);
        int kb = ck * 32 + 8 * (ln2 >> 4);
        for (int kk = 0; kk < 16; ++kk) {
            int K = kb + kk + (kk & 8);
            float v = (K < 16)  ? W_ih[n * 16 + K]
                    : (K < 80)  ? W_hh[n * 64 + (K - 16)]
                    : (K == 80) ? (b_ih[n] + b_hh[n]) : 0.0f;
            wfrag[it * 16 + kk] = (_Float16)v;
        }
    }
    __syncthreads();

    // ---------------- per-wave state init ----------------
    const int wv  = tid >> 5;
    const int ln  = tid & 31;
    const int hlf = ln >> 4;     // 0/1: wave half (K half / M half / rel component)
    const int col = ln & 15;     // N index in tile == local batch row for rel path
    const int rowbase = (blockIdx.x * WAVES_ + wv) * 16;
    const int grow = rowbase + col;
    _Float16* hb = &hbuf[wv][0];

    // pad region: K 80..95 -> positions 72..79 (K 80..87) and 88..95 (K 88..95)
    if (hlf == 0) {
        for (int p = 72; p < 80; ++p)
            hb[col * KTOT_ + p] = (_Float16)((p == 72) ? 1.0f : 0.0f); // pos 72 == K 80 == bias one
    } else {
        for (int p = 88; p < 96; ++p) hb[col * KTOT_ + p] = (_Float16)0.0f;
    }
    // h0 -> K [16,80) at permuted positions (this lane covers K 16+32*hlf .. +31)
    {
        const float4* h4 = (const float4*)(h0 + (size_t)grow * 64 + hlf * 32);
        float hv[32];
        for (int q = 0; q < 8; ++q) {
            float4 v = h4[q];
            hv[q * 4 + 0] = v.x; hv[q * 4 + 1] = v.y; hv[q * 4 + 2] = v.z; hv[q * 4 + 3] = v.w;
        }
        for (int k = 0; k < 32; ++k)
            hb[col * KTOT_ + posOf(16 + hlf * 32 + k)] = (_Float16)hv[k];
    }
    // x0 = lrelu(traj_rel @ W_se^T + b_se): K 0..7 -> pos 0..7 (hlf0), K 8..15 -> pos 16..23 (hlf1)
    {
        float tr  = traj_rel[(size_t)grow * 2 + hlf];
        float trp = __shfl_xor(tr, 16);
        float r0 = hlf ? trp : tr, r1 = hlf ? tr : trp;
        v8h xv;
        for (int j = 0; j < 8; ++j) {
            int k = hlf * 8 + j;
            xv[j] = (_Float16)lrelu(r0 * Wsel[k * 2] + r1 * Wsel[k * 2 + 1] + bsel[k]);
        }
        *(v8h*)(hb + col * KTOT_ + 16 * hlf) = xv;
    }
    // c in WMMA C/D layout: cst[t][r] = c[row = r + 8*hlf][hid = 16t + col]
    float cst[4][8];
    for (int t = 0; t < 4; ++t)
        for (int r = 0; r < 8; ++r)
            cst[t][r] = c0[(size_t)(rowbase + r + 8 * hlf) * 64 + 16 * t + col];
    float ap = traj_abs[(size_t)grow * 2 + hlf];     // abs_pos component hlf of row col
    float gt = 0.f;                                  // goal term of rel (constant per row)
    for (int j = 0; j < 64; ++j)
        gt += goals[(size_t)grow * 64 + j] * Mgl[j * 2 + hlf];
    // loop-invariant store positions for new-h column (hid tile t, this lane's col)
    int hpos[4];
    for (int t = 0; t < 4; ++t) hpos[t] = posOf(16 + 16 * t + col);
    __syncthreads();

    // ---------------- 30 recurrent steps ----------------
    #pragma unroll 1
    for (int s = 0; s < SEQ_LEN_; ++s) {
        // A fragments: one contiguous 32B v16h load per chunk (permuted layout)
        v16h a[3];
        #pragma unroll
        for (int ck = 0; ck < 3; ++ck)
            a[ck] = *(const v16h*)(hb + col * KTOT_ + ck * 32 + 16 * hlf);
        // gates per hidden 16-tile: i/f/g/o tiles {t, 4+t, 8+t, 12+t}; bias comes via K=80
        #pragma unroll
        for (int t = 0; t < 4; ++t) {
            v8f acc[4];
            #pragma unroll
            for (int g = 0; g < 4; ++g) {
                #pragma unroll
                for (int e = 0; e < 8; ++e) acc[g][e] = 0.0f;
            }
            #pragma unroll
            for (int ck = 0; ck < 3; ++ck) {
                #pragma unroll
                for (int g = 0; g < 4; ++g) {
                    int nt = g * 4 + t;
                    const v16h* wp = (const v16h*)&wfrag[((nt * 3 + ck) * 32 + ln) * 16];
                    acc[g] = __builtin_amdgcn_wmma_f32_16x16x32_f16(
                        false, a[ck], false, *wp, (short)0, acc[g], false, false);
                }
            }
            #pragma unroll
            for (int r = 0; r < 8; ++r) {
                float iv = fast_sig(acc[0][r]);
                float fv = fast_sig(acc[1][r]);
                float gv = fast_tanh(acc[2][r]);
                float ov = fast_sig(acc[3][r]);
                float c  = fv * cst[t][r] + iv * gv;
                cst[t][r] = c;
                float h  = ov * fast_tanh(c);
                hb[(r + 8 * hlf) * KTOT_ + hpos[t]] = (_Float16)h;
            }
        }
        __syncthreads();
        // rel = h @ W1^T + abs_pos @ M2 + goal-term + const   (component hlf, row col)
        // straight scan of storage positions 8..87; W1P is pre-permuted (zeros on x/bias/pad)
        float hd = 0.f;
        const _Float16* hr = hb + col * KTOT_ + 8;
        #pragma unroll
        for (int kb = 0; kb < 10; ++kb) {
            v8h hv = *(const v8h*)(hr + kb * 8);
            #pragma unroll
            for (int e = 0; e < 8; ++e)
                hd += (float)hv[e] * W1P[hlf * 80 + kb * 8 + e];
        }
        float app = __shfl_xor(ap, 16);
        float a0 = hlf ? app : ap, a1 = hlf ? ap : app;
        float rel = hd + a0 * M2l[hlf] + a1 * M2l[2 + hlf] + gt + crell[hlf];
        out[((size_t)s * Btot + grow) * 2 + hlf] = rel;
        ap += rel;
        // x = lrelu(rel @ W_se^T + b_se): single v8h store per lane
        float relp = __shfl_xor(rel, 16);
        float r0 = hlf ? relp : rel, r1 = hlf ? rel : relp;
        v8h xv;
        #pragma unroll
        for (int j = 0; j < 8; ++j) {
            int k = hlf * 8 + j;
            xv[j] = (_Float16)lrelu(r0 * Wsel[k * 2] + r1 * Wsel[k * 2 + 1] + bsel[k]);
        }
        *(v8h*)(hb + col * KTOT_ + 16 * hlf) = xv;
        __syncthreads();
    }
}

extern "C" void kernel_launch(void* const* d_in, const int* in_sizes, int n_in,
                              void* d_out, int out_size, void* d_ws, size_t ws_size,
                              hipStream_t stream) {
    (void)n_in; (void)out_size; (void)d_ws; (void)ws_size;
    const int B = in_sizes[0] / 2;          // traj_abs is (1, B, 2)
    dim3 grid(B / (16 * WAVES_));           // one wave per 16 rows, 4 waves per block
    goal_lstm_kernel<<<grid, 128, 0, stream>>>(
        (const float*)d_in[0],  (const float*)d_in[1],
        (const float*)d_in[2],  (const float*)d_in[3],
        (const float*)d_in[4],
        (const float*)d_in[5],  (const float*)d_in[6],
        (const float*)d_in[7],  (const float*)d_in[8],
        (const float*)d_in[9],  (const float*)d_in[10],
        (const float*)d_in[11], (const float*)d_in[12],
        (const float*)d_in[13], (const float*)d_in[14],
        (const float*)d_in[15], (const float*)d_in[16],
        (float*)d_out, B);
}
